// HOPE_SSM_62843961475575
// MI455X (gfx1250) — matrix operand order repaired and checked by reading, not compiled, not run on hardware
//
#include <hip/hip_runtime.h>
#include <hip/hip_bf16.h>
#include <math.h>

// Problem constants from the reference
#define BB   16
#define DM   512
#define LL   2048
#define H2   1024
#define NH   64
#define LK   1024          // L/2
#define NPOS (BB * LL)     // 32768 GEMM columns

typedef __attribute__((ext_vector_type(16))) __bf16 v16bf;
typedef __attribute__((ext_vector_type(8)))  __bf16 v8bf;
typedef __attribute__((ext_vector_type(8)))  float  v8f;

union FragAB { v16bf v; v8bf h[2]; };

// ---------------------------------------------------------------------------
// Kernel 1: build combined frequency response k[c][l] (c<512, l<2048), f32 cplx
//   l <  LK : K[c][l]            (h = c)
//   l >= LK : K[c+512][2LK-1-l]  (flipped second half)
// K(h,lk) = sum_n (Hr+iHi)(h,n) * exp(-i*theta(h,lk)*(n+1))
// ---------------------------------------------------------------------------
__global__ __launch_bounds__(256) void hope_filter_kernel(
    const float* __restrict__ log_dt, const float* __restrict__ Hr,
    const float* __restrict__ Hi, float2* __restrict__ kf)
{
    int idx = blockIdx.x * 256 + threadIdx.x;          // 512*2048 total
    int c = idx >> 11;
    int l = idx & (LL - 1);
    int h, lk;
    if (l < LK) { h = c;       lk = l;           }
    else        { h = c + DM;  lk = 2 * LK - 1 - l; }

    float dt = __expf(log_dt[h]);
    float ang = 6.28318530717958647692f * (float)lk / (float)LK;
    float sn, cs;
    __sincosf(ang, &sn, &cs);
    // sd = ((1+dt)s + dt-1) / ((dt-1)s + dt+1)
    float nr = (1.f + dt) * cs + (dt - 1.f);
    float ni = (1.f + dt) * sn;
    float dr = (dt - 1.f) * cs + (dt + 1.f);
    float di = (dt - 1.f) * sn;
    float inv = 1.f / (dr * dr + di * di);
    float sdr = (nr * dr + ni * di) * inv;
    float sdi = (ni * dr - nr * di) * inv;
    float theta = atan2f(sdi, sdr);

    // e = exp(-i*theta); p starts at e (n+1 = 1)
    float es, ec;
    __sincosf(theta, &es, &ec);
    float er = ec, ei = -es;
    float pr = er, pi = ei;
    const float* hr = Hr + h * NH;
    const float* hi = Hi + h * NH;
    float ar = 0.f, ai = 0.f;
#pragma unroll 8
    for (int n = 0; n < NH; ++n) {
        float xr = hr[n], xi = hi[n];
        ar += xr * pr - xi * pi;
        ai += xr * pi + xi * pr;
        float tpr = pr * er - pi * ei;
        float tpi = pr * ei + pi * er;
        pr = tpr; pi = tpi;
    }
    kf[idx] = make_float2(ar, ai);
}

// ---------------------------------------------------------------------------
// Kernel 2: W (f32, H2 x DM) -> bf16 workspace (row major, contiguous K runs)
// ---------------------------------------------------------------------------
__global__ __launch_bounds__(256) void hope_wcvt_kernel(
    const float* __restrict__ W, __bf16* __restrict__ Wb)
{
    int idx = blockIdx.x * 256 + threadIdx.x;          // H2*DM total
    Wb[idx] = (__bf16)W[idx];
}

// ---------------------------------------------------------------------------
// Kernel 3: per-sequence FFT -> multiply by k -> IFFT -> +u*D -> GELU -> bf16
// One workgroup (256 thr) per (b,c) sequence; 2048-pt radix-2 FFT in LDS with
// an LDS twiddle table (1024 sincos per block instead of ~22K).
// Output written COALESCED in natural [b][c][l] order (bf16); a separate
// transpose kernel produces the [pos][c] layout the GEMM wants.
// ---------------------------------------------------------------------------
__global__ __launch_bounds__(256) void hope_fftmul_kernel(
    const float* __restrict__ u, const float2* __restrict__ kf,
    const float* __restrict__ Dv, __bf16* __restrict__ Ybc)
{
    __shared__ float sre[LL];
    __shared__ float sim[LL];
    __shared__ float twr[LK];   // T[k] = exp(-2*pi*i*k/2048), k < 1024
    __shared__ float twi[LK];

    int tid = threadIdx.x;
    int seq = blockIdx.x;                 // b*DM + c
    int b   = seq / DM;
    int c   = seq - b * DM;
    size_t base = (size_t)seq * LL;
    float Dc = Dv[c];

    // twiddle table
    for (int k = tid; k < LK; k += 256) {
        float s, cc;
        __sincosf(-6.28318530717958647692f * (float)k / (float)LL, &s, &cc);
        twr[k] = cc;
        twi[k] = s;
    }
    // load with bit reversal (11 bits): x[i] = u[rev(i)]
    for (int i = tid; i < LL; i += 256) {
        int j = (int)(__brev((unsigned)i) >> 21);
        sre[i] = u[base + j];
        sim[i] = 0.f;
    }
    __syncthreads();

    // forward FFT: twiddle exp(-2*pi*i*j/len) = T[j * (2048/len)]
    for (int len = 2, mul = LK; len <= LL; len <<= 1, mul >>= 1) {
        int hl = len >> 1;
        for (int t = tid; t < (LL >> 1); t += 256) {
            int j  = t & (hl - 1);
            int i0 = 2 * t - j;
            int i1 = i0 + hl;
            float cc = twr[j * mul];
            float s  = twi[j * mul];
            float vr = sre[i1], vi = sim[i1];
            float tr = vr * cc - vi * s;
            float ti = vr * s + vi * cc;
            float ur = sre[i0], ui = sim[i0];
            sre[i0] = ur + tr; sim[i0] = ui + ti;
            sre[i1] = ur - tr; sim[i1] = ui - ti;
        }
        __syncthreads();
    }

    // multiply by filter
    const float2* kc = kf + (size_t)c * LL;
    for (int i = tid; i < LL; i += 256) {
        float2 kv = kc[i];
        float xr = sre[i], xi = sim[i];
        sre[i] = xr * kv.x - xi * kv.y;
        sim[i] = xr * kv.y + xi * kv.x;
    }
    __syncthreads();

    // in-place bit-reverse permutation (involution: swap pairs i<j)
    for (int i = tid; i < LL; i += 256) {
        int j = (int)(__brev((unsigned)i) >> 21);
        if (i < j) {
            float tr = sre[i]; sre[i] = sre[j]; sre[j] = tr;
            float ti = sim[i]; sim[i] = sim[j]; sim[j] = ti;
        }
    }
    __syncthreads();

    // inverse FFT: conjugate twiddles, scale 1/L at the end
    for (int len = 2, mul = LK; len <= LL; len <<= 1, mul >>= 1) {
        int hl = len >> 1;
        for (int t = tid; t < (LL >> 1); t += 256) {
            int j  = t & (hl - 1);
            int i0 = 2 * t - j;
            int i1 = i0 + hl;
            float cc =  twr[j * mul];
            float s  = -twi[j * mul];
            float vr = sre[i1], vi = sim[i1];
            float tr = vr * cc - vi * s;
            float ti = vr * s + vi * cc;
            float ur = sre[i0], ui = sim[i0];
            sre[i0] = ur + tr; sim[i0] = ui + ti;
            sre[i1] = ur - tr; sim[i1] = ui - ti;
        }
        __syncthreads();
    }

    // y = conv.real/L + u*D, exact GELU, coalesced bf16 store at [b][c][l]
    const float invL = 1.0f / (float)LL;
    for (int i = tid; i < LL; i += 256) {
        float yr = sre[i] * invL + u[base + i] * Dc;
        float g  = 0.5f * yr * (1.f + erff(yr * 0.70710678118654752f));
        Ybc[base + i] = (__bf16)g;
    }
}

// ---------------------------------------------------------------------------
// Kernel 3b: transpose Ybc[b][c][l] (bf16) -> Yb[pos = b*L + l][c] (bf16).
// 32x32 LDS tiles (padded) so both global read and write are coalesced.
// ---------------------------------------------------------------------------
__global__ __launch_bounds__(256) void hope_transpose_kernel(
    const __bf16* __restrict__ Ybc, __bf16* __restrict__ Yb)
{
    __shared__ __bf16 t[32][33];

    int bt = blockIdx.x;                 // 16 b * 16 c-tiles * 64 l-tiles
    int lt = bt & 63;
    int ct = (bt >> 6) & 15;
    int b  = bt >> 10;
    int l0 = lt << 5;
    int c0 = ct << 5;

    int tx = threadIdx.x & 31;           // fast axis
    int ty = threadIdx.x >> 5;           // 0..7

#pragma unroll
    for (int q = 0; q < 4; ++q) {
        int cr = ty + q * 8;
        t[cr][tx] = Ybc[((size_t)(b * DM + c0 + cr) << 11) + l0 + tx];
    }
    __syncthreads();
#pragma unroll
    for (int q = 0; q < 4; ++q) {
        int lr = ty + q * 8;
        Yb[(size_t)(b * LL + l0 + lr) * DM + c0 + tx] = t[tx][lr];
    }
}

// ---------------------------------------------------------------------------
// Kernel 4: WMMA GEMM + fused bias + GLU.
//   z = W(1024x512) * Y(512x32768);  out[o,pos] = (z[o]+b[o])*sigmoid(z[o+512]+b[o+512])
// Each wave owns a 16(M) x 64(N) output slab: the two A fragments (a & g rows
// of W) are reused across 4 B fragments -> 8 WMMAs per 6KB of fragment fetch
// (~2x the arithmetic intensity of a 16x16 tile). K=512 swept in 32-wide
// bf16 WMMA steps with f32 accumulation.
// ---------------------------------------------------------------------------
__global__ __launch_bounds__(256) void hope_gemm_glu_kernel(
    const __bf16* __restrict__ Wb, const __bf16* __restrict__ Yb,
    const float* __restrict__ bias, float* __restrict__ out)
{
    int tid  = threadIdx.x;
    int wave = tid >> 5;
    int lane = tid & 31;
    int lr   = lane & 15;
    int half = lane >> 4;

    int tile   = blockIdx.x * 8 + wave;       // 32 m-tiles * 512 n-supertiles
    int m_tile = tile & 31;                   // o in [0,512) by 16
    int n_tile = tile >> 5;                   // pos by 64
    int o_base   = m_tile << 4;
    int pos_base = n_tile << 6;

    const __bf16* Wra = Wb + (size_t)(o_base + lr) * DM;        // 'a' rows
    const __bf16* Wrg = Wb + (size_t)(o_base + DM + lr) * DM;   // 'g' rows
    const __bf16* Yr  = Yb + (size_t)(pos_base + lr) * DM;      // B rows

    v8f accA[4] = {};
    v8f accG[4] = {};

    for (int kk = 0; kk < DM; kk += 32) {
        // A 16x32 bf16 layout: lanes 0-15 row M=lr K={0..7,16..23}+kk,
        //                      lanes 16-31 same rows K={8..15,24..31}+kk
        int ka = kk + (half ? 8 : 0);
        FragAB fa, fg;
        fa.h[0] = *(const v8bf*)(Wra + ka);
        fa.h[1] = *(const v8bf*)(Wra + ka + 16);
        fg.h[0] = *(const v8bf*)(Wrg + ka);
        fg.h[1] = *(const v8bf*)(Wrg + ka + 16);

        __builtin_prefetch(Yr + kk + 64, 0, 1);   // next K panel

        // B 32x16 bf16 layout: lane holds column N=lr, K={0..7,16..23}+kk (lanes 0-15), K={8..15,24..31}+kk (lanes 16-31)
        int kb = kk + (half ? 8 : 0);
#pragma unroll
        for (int s = 0; s < 4; ++s) {
            FragAB fb;
            const __bf16* Ys = Yr + (size_t)(s * 16) * DM;
            fb.h[0] = *(const v8bf*)(Ys + kb);
            fb.h[1] = *(const v8bf*)(Ys + kb + 16);
            accA[s] = __builtin_amdgcn_wmma_f32_16x16x32_bf16(
                false, fa.v, false, fb.v, (short)0, accA[s], false, false);
            accG[s] = __builtin_amdgcn_wmma_f32_16x16x32_bf16(
                false, fg.v, false, fb.v, (short)0, accG[s], false, false);
        }
    }

    // Epilogue: C/D layout -> lane<16: M=r, N=lane ; lane>=16: M=r+8, N=lane-16
#pragma unroll
    for (int s = 0; s < 4; ++s) {
        int pos  = pos_base + s * 16 + lr;
        int bidx = pos >> 11;            // pos / L
        int l    = pos & (LL - 1);
#pragma unroll
        for (int r = 0; r < 8; ++r) {
            int M = r + (half ? 8 : 0);
            int o = o_base + M;
            float av = accA[s][r] + bias[o];
            float gv = accG[s][r] + bias[o + DM];
            float sg = 1.f / (1.f + __expf(-gv));
            out[((size_t)(bidx * DM + o) << 11) + l] = av * sg;
        }
    }
}

// ---------------------------------------------------------------------------
// Host-side launch
// ---------------------------------------------------------------------------
extern "C" void kernel_launch(void* const* d_in, const int* in_sizes, int n_in,
                              void* d_out, int out_size, void* d_ws, size_t ws_size,
                              hipStream_t stream) {
    (void)in_sizes; (void)n_in; (void)out_size; (void)ws_size;
    const float* u      = (const float*)d_in[0];
    const float* log_dt = (const float*)d_in[1];
    const float* Hr     = (const float*)d_in[2];
    const float* Hi     = (const float*)d_in[3];
    const float* Dv     = (const float*)d_in[4];
    const float* W      = (const float*)d_in[5];
    const float* bias   = (const float*)d_in[6];
    float* out          = (float*)d_out;

    char* ws = (char*)d_ws;
    float2* kf  = (float2*)ws;                           //  8 MB: 512*2048 cplx f32
    __bf16* Wb  = (__bf16*)(ws + (8u  << 20));           //  1 MB: 1024*512 bf16
    __bf16* Ybc = (__bf16*)(ws + (9u  << 20));           // 32 MB: y in [b][c][l] bf16
    __bf16* Yb  = (__bf16*)(ws + (41u << 20));           // 32 MB: y in [pos][c] bf16

    // 1) frequency-domain filter
    hope_filter_kernel<<<(DM * LL) / 256, 256, 0, stream>>>(log_dt, Hr, Hi, kf);
    // 2) W -> bf16
    hope_wcvt_kernel<<<(H2 * DM) / 256, 256, 0, stream>>>(W, Wb);
    // 3) FFT * k -> IFFT -> skip + GELU -> bf16, coalesced natural layout
    hope_fftmul_kernel<<<BB * DM, 256, 0, stream>>>(u, kf, Dv, Ybc);
    // 3b) transpose to GEMM layout [pos][c]
    hope_transpose_kernel<<<BB * 16 * 64, 256, 0, stream>>>(Ybc, Yb);
    // 4) WMMA GEMM + bias + GLU: 32 m-tiles x 512 n-supertiles, 8 waves/block
    hope_gemm_glu_kernel<<<(32 * (NPOS / 64)) / 8, 256, 0, stream>>>(Wb, Yb, bias, out);
}
